// DecoderFull_6416681141094
// MI455X (gfx1250) — compile-verified, matrix operand order checked by reading, not run on hardware
//
#include <hip/hip_runtime.h>
#include <hip/hip_bf16.h>

typedef __attribute__((ext_vector_type(16))) _Float16 v16h;
typedef __attribute__((ext_vector_type(8)))  _Float16 v8h;
typedef __attribute__((ext_vector_type(8)))  float    v8f;

#define DEC_WS 20480   // per-decoder workspace stride (bytes)
#define LDS_BYTES 18560
#define NW 4           // waves per block

// ---------------------------------------------------------------------------
// CDNA5 transcendental tanh (V_TANH_F32): builtin confirmed present, the
// backend models the TRANS hazard itself.
#if __has_builtin(__builtin_amdgcn_tanhf)
__device__ __forceinline__ float tanh_f(float x) {
  return __builtin_amdgcn_tanhf(x);
}
#elif __has_builtin(__builtin_amdgcn_tanh_f32)
__device__ __forceinline__ float tanh_f(float x) {
  return __builtin_amdgcn_tanh_f32(x);
}
#else
__device__ __forceinline__ float tanh_f(float x) {
  float r;
  asm("v_tanh_f32 %0, %1\n\tv_nop" : "=v"(r) : "v"(x));
  return r;
}
#endif
// Sigmoid on a pre-scaled argument: weights of i/f/o gates carry the 0.5x,
// so sigmoid(x) = fma(0.5, tanh(y), 0.5) with y = 0.5x already in the gemm.
__device__ __forceinline__ float sigm_pre(float y) {
  return __builtin_fmaf(0.5f, tanh_f(y), 0.5f);
}

// ---------------------------------------------------------------------------
// Precompute: per decoder, build f16 weight packs + folded weights in ws.
//   W_eff = W_hh + W_ih @ W_out      (prev-feedback folded into recurrence)
//   b_eff = b_ih + b_hh + W_ih @ b_out
// Rows of sigmoid gates (i:0-31, f:32-63, o:96-127) are pre-scaled by 0.5.
struct DecP {
  const float *Wih, *Whh, *bih, *bhh, *Wout, *bout;
  int din, dout;
};
struct DecP3 { DecP d[3]; };

__global__ void precompute_kernel(DecP3 P, unsigned char* ws) {
  const int dec = blockIdx.x;
  const DecP p = P.d[dec];
  unsigned char* w = ws + dec * DEC_WS;
  _Float16* WhhH  = (_Float16*)(w);            // 128x32 f16 (gate-scaled)
  _Float16* WeffH = (_Float16*)(w + 8192);     // 128x32 f16 (gate-scaled)
  _Float16* WoutH = (_Float16*)(w + 16384);    // 16x32  f16 (zero padded)
  float* bias0  = (float*)(w + 17408);         // 128 f32 (gate-scaled)
  float* biasE  = (float*)(w + 17920);         // 128 f32 (gate-scaled)
  float* bout16 = (float*)(w + 18432);         // 16  f32 (zero padded)
  const int g = threadIdx.x;                   // 0..127 = gate row
  const float scl = (g < 64 || g >= 96) ? 0.5f : 1.0f;  // i,f,o scaled

  float bi = p.bih[g] + p.bhh[g];
  float be = bi;
  for (int j = 0; j < p.din; ++j) be += p.Wih[g * p.din + j] * p.bout[j];
  bias0[g] = bi * scl;
  biasE[g] = be * scl;
  for (int k = 0; k < 32; ++k) {
    float whh = p.Whh[g * 32 + k];
    float s = whh;
    for (int j = 0; j < p.din; ++j)
      s += p.Wih[g * p.din + j] * p.Wout[j * 32 + k];
    WhhH[g * 32 + k]  = (_Float16)(whh * scl);
    WeffH[g * 32 + k] = (_Float16)(s * scl);
  }
  if (g < 16) {
    bout16[g] = (g < p.dout) ? p.bout[g] : 0.0f;
    for (int k = 0; k < 32; ++k)
      WoutH[g * 32 + k] = (g < p.dout) ? (_Float16)p.Wout[g * 32 + k]
                                       : (_Float16)0.0f;
  }
}

// ---------------------------------------------------------------------------
// A-operand loader (from LDS): 16x32 f16 tile in WMMA A layout.
// lane<16 holds row m with K = {0..7, 16..23}; lane>=16 holds K = {8..15, 24..31}.
__device__ __forceinline__ v16h load_a_tile(const _Float16* W, int row, int o) {
  const _Float16* r = W + row * 32;
  v8h lo = *(const v8h*)(r + o);
  v8h hi = *(const v8h*)(r + 16 + o);
  v16h a;
#pragma unroll
  for (int e = 0; e < 8; ++e) { a[e] = lo[e]; a[8 + e] = hi[e]; }
  return a;
}

__device__ __forceinline__ v8f wmma_step(const v16h& a, const v16h& b,
                                         const v8f& c) {
  return __builtin_amdgcn_wmma_f32_16x16x32_f16(
      false, a, false, b, (short)0, c, false, false);
}

// ---------------------------------------------------------------------------
// Main kernel: one wave per 16-sequence tile per decoder; 4 waves/block share
// one decoder's weight pack staged in LDS (320KB/WGP on CDNA5).
// gates^T(128x16) = W(128x32) x h^T(32x16) : 8 WMMAs; pred^T: 1 WMMA.
// Weight operands come from LDS (register-kept or ds_load — never VMEM in the
// steady-state loop); recurrence state (h, c) never leaves VGPRs.
__global__ __launch_bounds__(128, 1) void lstm3_kernel(
    const float* __restrict__ x,
    const int* __restrict__ len_lidar,
    const int* __restrict__ len_camera,
    const int* __restrict__ Tl_ptr,
    const int* __restrict__ Tc_ptr,
    float* __restrict__ out,
    const unsigned char* __restrict__ ws,
    int B)
{
  __shared__ __align__(16) unsigned char lds[LDS_BYTES];

  const int tid  = threadIdx.x;
  const int wave = tid >> 5;
  const int lane = tid & 31;
  const int dec  = blockIdx.y;
  const int tile = blockIdx.x * NW + wave;

  // Stage this decoder's weight pack into LDS (all threads participate,
  // barrier before any possible early exit).
  {
    const uint4* src = (const uint4*)(ws + dec * DEC_WS);
    uint4* dst = (uint4*)lds;
    for (int i = tid; i < LDS_BYTES / 16; i += 128) dst[i] = src[i];
  }
  __syncthreads();

  if (tile * 16 >= B) return;                 // wave-uniform: EXEC stays all-1s
  const int base = tile * 16;
  const int m = lane & 15;                    // batch column / weight row
  const int o = (lane < 16) ? 0 : 8;          // K/hidden half-offset

  const _Float16* WhhH  = (const _Float16*)(lds);
  const _Float16* WeffH = (const _Float16*)(lds + 8192);
  const _Float16* WoutH = (const _Float16*)(lds + 16384);
  const float* bias0  = (const float*)(lds + 17408);
  const float* biasE  = (const float*)(lds + 17920);
  const float* bout16 = (const float*)(lds + 18432);

  const int Tl = Tl_ptr[0];
  const int Tc = Tc_ptr[0];
  const int T    = (dec == 0) ? Tl : Tc;
  const int dout = (dec == 2) ? 4 : 3;
  const int* lens = (dec == 0) ? len_lidar : len_camera;
  const int mylen = lens[base + m];

  long outOff = 0;
  if (dec >= 1) outOff += (long)B * Tl * 3;
  if (dec >= 2) outOff += (long)B * Tc * 3;
  float* __restrict__ op0 = out + outOff;

  // h0 = c0 = enc. Load per-lane hidden halves straight into D-layout c and
  // B-layout h (same per-lane data, different packing).
  const float* xr = x + (long)(base + m) * 32 + o;
  v8f c0 = *(const v8f*)(xr);        // hidden o..o+7
  v8f c1 = *(const v8f*)(xr + 16);   // hidden o+16..o+23
  v16h bh;
#pragma unroll
  for (int e = 0; e < 8; ++e) {
    bh[e]     = (_Float16)c0[e];
    bh[8 + e] = (_Float16)c1[e];
  }

  // Loop-invariant operands (LDS-backed).
  v16h wO = load_a_tile(WoutH, m, o);          // padded W_out as A
  v8f  cO = *(const v8f*)(bout16 + o);         // b_out in D layout (C operand)

  v16h wA[8];
  v8f  cb[8];
#pragma unroll
  for (int tq = 0; tq < 8; ++tq) {             // step 0 uses plain W_hh
    wA[tq] = load_a_tile(WhhH, tq * 16 + m, o);
    cb[tq] = *(const v8f*)(bias0 + tq * 16 + o);
  }

  auto step_store = [&](int t) {
    // ---- forget gate: consume immediately, c *= sigmoid(f)
    {
      v8f f0 = wmma_step(wA[2], bh, cb[2]);
      v8f f1 = wmma_step(wA[3], bh, cb[3]);
#pragma unroll
      for (int e = 0; e < 8; ++e) {
        c0[e] *= sigm_pre(f0[e]);
        c1[e] *= sigm_pre(f1[e]);
      }
    }
    // ---- input gate * candidate: c += sigmoid(i) * tanh(g)
    {
      v8f i0 = wmma_step(wA[0], bh, cb[0]);
      v8f i1 = wmma_step(wA[1], bh, cb[1]);
      v8f g0 = wmma_step(wA[4], bh, cb[4]);
      v8f g1 = wmma_step(wA[5], bh, cb[5]);
#pragma unroll
      for (int e = 0; e < 8; ++e) {
        c0[e] = __builtin_fmaf(sigm_pre(i0[e]), tanh_f(g0[e]), c0[e]);
        c1[e] = __builtin_fmaf(sigm_pre(i1[e]), tanh_f(g1[e]), c1[e]);
      }
    }
    // ---- output gate: h = sigmoid(o) * tanh(c); repack as next B operand
    {
      v8f o0 = wmma_step(wA[6], bh, cb[6]);   // still uses OLD bh
      v8f o1 = wmma_step(wA[7], bh, cb[7]);
#pragma unroll
      for (int e = 0; e < 8; ++e) {
        bh[e]     = (_Float16)(sigm_pre(o0[e]) * tanh_f(c0[e]));
        bh[8 + e] = (_Float16)(sigm_pre(o1[e]) * tanh_f(c1[e]));
      }
    }
    // pred^T = W_out_pad x h^T + b_out : lanes 0..15 hold one pred vector each
    v8f p = wmma_step(wO, bh, cO);
    if (lane < 16) {
      float msk = (t < mylen) ? 1.0f : 0.0f;
      float* dst = op0 + ((long)(base + m) * T + t) * dout;
      dst[0] = p[0] * msk;
      dst[1] = p[1] * msk;
      dst[2] = p[2] * msk;
      if (dout == 4) dst[3] = p[3] * msk;
    }
  };

  step_store(0);

  // Switch to folded weights: prev-feedback matmul eliminated for t>=1.
#pragma unroll
  for (int tq = 0; tq < 8; ++tq) {
    wA[tq] = load_a_tile(WeffH, tq * 16 + m, o);
    cb[tq] = *(const v8f*)(biasE + tq * 16 + o);
  }
#pragma clang loop unroll(disable)
  for (int t = 1; t < T; ++t) step_store(t);
}

// ---------------------------------------------------------------------------
extern "C" void kernel_launch(void* const* d_in, const int* in_sizes, int n_in,
                              void* d_out, int out_size, void* d_ws, size_t ws_size,
                              hipStream_t stream) {
  (void)n_in; (void)out_size; (void)ws_size;
  const float* x = (const float*)d_in[0];
  DecP3 P;
  P.d[0] = DecP{(const float*)d_in[1],  (const float*)d_in[2],
                (const float*)d_in[3],  (const float*)d_in[4],
                (const float*)d_in[5],  (const float*)d_in[6],
                in_sizes[1] / 128, in_sizes[5] / 32};
  P.d[1] = DecP{(const float*)d_in[7],  (const float*)d_in[8],
                (const float*)d_in[9],  (const float*)d_in[10],
                (const float*)d_in[11], (const float*)d_in[12],
                in_sizes[7] / 128, in_sizes[11] / 32};
  P.d[2] = DecP{(const float*)d_in[13], (const float*)d_in[14],
                (const float*)d_in[15], (const float*)d_in[16],
                (const float*)d_in[17], (const float*)d_in[18],
                in_sizes[13] / 128, in_sizes[17] / 32};
  const int* len_l = (const int*)d_in[19];
  const int* len_c = (const int*)d_in[20];
  const int* Tl = (const int*)d_in[22];
  const int* Tc = (const int*)d_in[23];
  const int B = in_sizes[0] / 32;   // x is (B, H=32, 1, 1)

  precompute_kernel<<<3, 128, 0, stream>>>(P, (unsigned char*)d_ws);

  const int tiles = B / 16;
  dim3 grid((tiles + NW - 1) / NW, 3);
  lstm3_kernel<<<grid, 128, 0, stream>>>(x, len_l, len_c, Tl, Tc,
                                         (float*)d_out,
                                         (const unsigned char*)d_ws, B);
}
